// Sea_Attention_43276090474579
// MI455X (gfx1250) — matrix operand order repaired and checked by reading, not compiled, not run on hardware
//
#include <hip/hip_runtime.h>

typedef __attribute__((ext_vector_type(16))) __bf16 v16bf;
typedef __attribute__((ext_vector_type(8)))  float  v8f;

#define AST 40   // LDS tile row stride in ushorts (80B: 16B-aligned for async B128, bank-spread)

__device__ __forceinline__ unsigned short f32_to_bf16(float f){
  unsigned int u = __float_as_uint(f);
  u += 0x7FFFu + ((u >> 16) & 1u);           // round-to-nearest-even
  return (unsigned short)(u >> 16);
}

// Flat LDS pointer -> 32-bit LDS byte offset (low 32 bits of the flat address).
__device__ __forceinline__ unsigned lds_off_of(const void* p){
  return (unsigned)(unsigned long long)p;
}

// Async DMA 16 bytes/lane from global to LDS (ASYNCcnt-tracked, wave-level).
__device__ __forceinline__ void async_load_b128(unsigned lds_byte_off, const void* gptr){
  unsigned long long ga = (unsigned long long)gptr;
  asm volatile("global_load_async_to_lds_b128 %0, %1, off"
               :: "v"(lds_byte_off), "v"(ga) : "memory");
}
__device__ __forceinline__ void wait_async0(){
  asm volatile("s_wait_asynccnt 0" ::: "memory");
}

union FragBF { unsigned int u[8]; v16bf v; };

// One K=32 step: wave computes two 16x16 f32 subtiles (same A row-strip, two B col-strips).
// LDS layouts: Aa[64][AST] bf16 row-major (m,k); Bt[64][AST] bf16 transposed (n,k).
__device__ __forceinline__ void wmma_step(const unsigned short* __restrict__ Aa,
                                          const unsigned short* __restrict__ Bt,
                                          int mr, int ncA, int lane,
                                          v8f& acc0, v8f& acc1)
{
  FragBF a, b0, b1;
  const int ml  = lane & 15;
  const int khA = (lane < 16) ? 0 : 8;       // A: lanes 16-31 hold K+8 halves
  const unsigned short* arow = Aa + (mr * 16 + ml) * AST;
#pragma unroll
  for (int j = 0; j < 8; ++j){
    int k0 = 2 * j + ((j >= 4) ? 8 : 0) + khA;   // per ISA 16-bit A 16x32 layout
    a.u[j] = *(const unsigned int*)(arow + k0);
  }
  const int kb = (lane < 16) ? 0 : 8;        // B: lanes 16-31 hold K+8 halves (same as A)
  const unsigned short* b0row = Bt + (ncA * 16 + ml) * AST;
  const unsigned short* b1row = b0row + 16 * AST;
#pragma unroll
  for (int j = 0; j < 8; ++j){
    int k0 = 2 * j + ((j >= 4) ? 8 : 0) + kb;
    b0.u[j] = *(const unsigned int*)(b0row + k0);
    b1.u[j] = *(const unsigned int*)(b1row + k0);
  }
  acc0 = __builtin_amdgcn_wmma_f32_16x16x32_bf16(false, a.v, false, b0.v, (short)0, acc0, false, false);
  acc1 = __builtin_amdgcn_wmma_f32_16x16x32_bf16(false, a.v, false, b1.v, (short)0, acc1, false, false);
}

// Generic batched GEMM: C[b][m][n] = (sum_k Abf[m][k]*B[b][k][n]) * scale[m] + bias[m]
// MODE 0: B is f32. MODE 1: B is f32 + relu. MODE 2: B is bf16 (ushort).
// Software-pipelined: A tile via async global->LDS DMA, B tile via regs (cvt/transpose),
// both double-buffered; prefetch of step s+1 overlaps WMMA of step s.
template<int MODE>
__global__ void __launch_bounds__(256) wmma_gemm(
    const unsigned short* __restrict__ Abf,
    const float* __restrict__ scale, const float* __restrict__ bias,
    const void* __restrict__ Bsrc,
    float* __restrict__ Cout,
    int M, int K, int N, long strideB, long strideC)
{
  __shared__ unsigned short Aa[2][64 * AST];
  __shared__ unsigned short Bt[2][64 * AST];
  const int b     = blockIdx.z;
  const int mBase = blockIdx.y * 64;
  const int nBase = blockIdx.x * 64;
  const int tid   = threadIdx.x;
  const int lane  = tid & 31, wave = tid >> 5;
  const int mr    = wave & 3;            // A row-strip 0..3
  const int ncA   = (wave >> 2) * 2;     // B col-strip pair: {0,1} or {2,3}
  v8f acc0 = {0,0,0,0,0,0,0,0}, acc1 = {0,0,0,0,0,0,0,0};
  const float*          Bf = (const float*)Bsrc          + (long)b * strideB;
  const unsigned short* Bh = (const unsigned short*)Bsrc + (long)b * strideB;

  // A tile = 64 rows x 32 bf16 = 256 chunks of 16B; chunk id == tid (8 waves x 32 lanes).
  const int ar = tid >> 2;               // row 0..63
  const int ac = (tid & 3) * 8;          // ushort offset of 16B chunk in row
  const unsigned aLds[2] = {
    lds_off_of(&Aa[0][0]) + (unsigned)(ar * AST + ac) * 2u,
    lds_off_of(&Aa[1][0]) + (unsigned)(ar * AST + ac) * 2u };
  const unsigned short* aG = Abf + (long)(mBase + ar) * K + ac;

  auto stageA = [&](int kt, int buf){
    async_load_b128(aLds[buf], (const void*)(aG + kt));
  };
  auto stageB = [&](int kt, int buf){
    unsigned short* Bd = Bt[buf];
    for (int idx = tid; idx < 2048; idx += 256){
      int kk = idx >> 6, j = idx & 63;
      unsigned short hv;
      if (MODE == 2){
        hv = Bh[(long)(kt + kk) * N + nBase + j];
      } else {
        float f = Bf[(long)(kt + kk) * N + nBase + j];
        if (MODE == 1) f = fmaxf(f, 0.f);
        hv = f32_to_bf16(f);
      }
      Bd[j * AST + kk] = hv;
    }
  };

  const int nsteps = K >> 5;
  stageA(0, 0);
  stageB(0, 0);
  wait_async0();
  __syncthreads();
  for (int s = 0; s < nsteps; ++s){
    const int cur = s & 1;
    if (s + 1 < nsteps){
      stageA((s + 1) << 5, cur ^ 1);
      stageB((s + 1) << 5, cur ^ 1);
    }
    wmma_step(Aa[cur], Bt[cur], mr, ncA, lane, acc0, acc1);
    wait_async0();
    __syncthreads();
  }

  const int nl   = lane & 15;
  const int mOff = (lane < 16) ? 0 : 8;  // f32 C: VGPR r -> M=r (lanes 0-15) / r+8 (16-31)
  float* Cb = Cout + (long)b * strideC;
#pragma unroll
  for (int r = 0; r < 8; ++r){
    int m = mBase + mr * 16 + r + mOff;
    float sc = scale[m], bi = bias[m];
    int n0 = nBase + ncA * 16 + nl;
    Cb[(long)m * N + n0]      = acc0[r] * sc + bi;
    Cb[(long)m * N + n0 + 16] = acc1[r] * sc + bi;
  }
}

// Pack weights to bf16, concat q/k/v scale+bias, interpolate positional encodings 16->64.
__global__ void __launch_bounds__(256) pack_kernel(
    const float* wq, const float* wk, const float* wv,
    const float* sq, const float* bq, const float* sk, const float* bk,
    const float* sv, const float* bv,
    const float* wpw, const float* wrow, const float* wcol, const float* wproj,
    const float* pe_rq, const float* pe_rk, const float* pe_cq, const float* pe_ck,
    unsigned short* Wqkv, unsigned short* Wpw, unsigned short* Wrow,
    unsigned short* Wcol, unsigned short* Wproj,
    float* sqkv, float* bqkv,
    float* peqr, float* pekr, float* peqc, float* pekc)
{
  long id = (long)blockIdx.x * 256 + threadIdx.x;
  if (id < 65536){
    int m = (int)(id >> 7), kk = (int)(id & 127);
    float v = (m < 128) ? wq[m * 128 + kk] : (m < 256) ? wk[(m - 128) * 128 + kk]
                                                       : wv[(m - 256) * 128 + kk];
    Wqkv[id] = f32_to_bf16(v); return;
  }
  id -= 65536;
  if (id < 65536){ Wpw[id]  = f32_to_bf16(wpw[id]);  return; } id -= 65536;
  if (id < 65536){ Wrow[id] = f32_to_bf16(wrow[id]); return; } id -= 65536;
  if (id < 65536){ Wcol[id] = f32_to_bf16(wcol[id]); return; } id -= 65536;
  if (id < 32768){ Wproj[id]= f32_to_bf16(wproj[id]);return; } id -= 32768;
  if (id < 512){
    int c = (int)id;
    sqkv[c] = (c < 128) ? sq[c] : (c < 256) ? sk[c - 128] : sv[c - 256];
    bqkv[c] = (c < 128) ? bq[c] : (c < 256) ? bk[c - 128] : bv[c - 256];
    return;
  }
  id -= 512;
  if (id < 32768){
    int which = (int)(id >> 13); int rem = (int)(id & 8191);
    int c = rem >> 6, i = rem & 63;
    const float* pe = (which == 0) ? pe_rq : (which == 1) ? pe_rk : (which == 2) ? pe_cq : pe_ck;
    float*      dst = (which == 0) ? peqr  : (which == 1) ? pekr  : (which == 2) ? peqc  : pekc;
    float coord = (i + 0.5f) * 0.25f - 0.5f;          // S=16, n=64
    coord = fminf(fmaxf(coord, 0.f), 15.f);
    int lo = (int)floorf(coord);
    int hi = lo + 1; if (hi > 15) hi = 15;
    float fr = coord - (float)lo;
    dst[c * 64 + i] = pe[c * 16 + lo] * (1.f - fr) + pe[c * 16 + hi] * fr;
  }
}

// Per (b,ch): row means (over W) and column means (over H) of a 64x64 map.
__global__ void __launch_bounds__(256) means_kernel(const float* __restrict__ qkv,
                                                    float* __restrict__ mrow,
                                                    float* __restrict__ mcol)
{
  const long bc = blockIdx.x;
  const float* src = qkv + bc * 4096;
  __shared__ float tile[64 * 65];
  for (int idx = threadIdx.x; idx < 4096; idx += 256){
    int h = idx >> 6, w = idx & 63;
    tile[h * 65 + w] = src[idx];
  }
  __syncthreads();
  int t = threadIdx.x;
  if (t < 64){
    float rs = 0.f, cs = 0.f;
    for (int i = 0; i < 64; ++i){ rs += tile[t * 65 + i]; cs += tile[i * 65 + t]; }
    mrow[bc * 64 + t] = rs * (1.f / 64.f);
    mcol[bc * 64 + t] = cs * (1.f / 64.f);
  }
}

// Depthwise 3x3 (pad=1) + BN + ReLU -> bf16 (ready as WMMA B-matrix for pwconv).
__global__ void __launch_bounds__(256) dwconv_relu_bf16(
    const float* __restrict__ qkv, const float* __restrict__ wdw,
    const float* __restrict__ sdw, const float* __restrict__ bdw,
    unsigned short* __restrict__ outp)
{
  const long bc = blockIdx.x;
  const int  ch = (int)(bc & 511);
  const float* src = qkv + bc * 4096;
  __shared__ float tile[66 * 66];
  for (int idx = threadIdx.x; idx < 66 * 66; idx += 256){
    int ty = idx / 66, tx = idx % 66;
    int h = ty - 1, w = tx - 1;
    float v = 0.f;
    if (h >= 0 && h < 64 && w >= 0 && w < 64) v = src[h * 64 + w];
    tile[idx] = v;
  }
  __syncthreads();
  float w9[9];
#pragma unroll
  for (int i = 0; i < 9; ++i) w9[i] = wdw[ch * 9 + i];
  const float s = sdw[ch], bi = bdw[ch];
  for (int p = threadIdx.x; p < 4096; p += 256){
    int h = p >> 6, w = p & 63;
    float sum = 0.f;
#pragma unroll
    for (int ky = 0; ky < 3; ++ky)
#pragma unroll
      for (int kx = 0; kx < 3; ++kx)
        sum += w9[ky * 3 + kx] * tile[(h + ky) * 66 + (w + kx)];
    outp[bc * 4096 + p] = f32_to_bf16(fmaxf(sum * s + bi, 0.f));
  }
}

// Axial attention (row: axis=0 over H using W-means; col: axis=1 over W using H-means).
// Writes pre-activation [b][head*32+d][pos] (ReLU fused into the following GEMM's B stage).
__global__ void __launch_bounds__(256) axial_attn(
    const float* __restrict__ mean_row, const float* __restrict__ mean_col,
    const float* __restrict__ peq_row, const float* __restrict__ pek_row,
    const float* __restrict__ peq_col, const float* __restrict__ pek_col,
    float* __restrict__ xxr_pre, float* __restrict__ xxc_pre)
{
  const int bh = blockIdx.x, b = bh >> 3, head = bh & 7;
  const int axis = blockIdx.y;
  const float* mean = axis ? mean_col : mean_row;
  const float* peq  = axis ? peq_col  : peq_row;
  const float* pek  = axis ? pek_col  : pek_row;
  float*       outp = axis ? xxc_pre  : xxr_pre;
  __shared__ float qr[64 * 16];
  __shared__ float kr[16 * 64];
  __shared__ float attn[64 * 65];
  __shared__ float vr[64 * 33];
  const int tid = threadIdx.x;
  for (int idx = tid; idx < 1024; idx += 256){
    int i = idx >> 4, kd = idx & 15;
    int ch = head * 16 + kd;
    qr[i * 16 + kd] = mean[((long)b * 512 + ch) * 64 + i] + peq[ch * 64 + i];
    kr[kd * 64 + i] = mean[((long)b * 512 + 128 + ch) * 64 + i] + pek[ch * 64 + i];
  }
  for (int idx = tid; idx < 2048; idx += 256){
    int j = idx >> 5, d = idx & 31;
    vr[j * 33 + d] = mean[((long)b * 512 + 256 + head * 32 + d) * 64 + j];
  }
  __syncthreads();
  for (int idx = tid; idx < 4096; idx += 256){
    int i = idx >> 6, j = idx & 63;
    float s = 0.f;
#pragma unroll
    for (int kd = 0; kd < 16; ++kd) s += qr[i * 16 + kd] * kr[kd * 64 + j];
    attn[i * 65 + j] = s * 0.25f;                       // SCALE = 16^-0.5
  }
  __syncthreads();
  if (tid < 64){
    float mx = -1e30f;
    for (int j = 0; j < 64; ++j) mx = fmaxf(mx, attn[tid * 65 + j]);
    float sum = 0.f;
    for (int j = 0; j < 64; ++j){ float e = __expf(attn[tid * 65 + j] - mx); attn[tid * 65 + j] = e; sum += e; }
    float inv = 1.f / sum;
    for (int j = 0; j < 64; ++j) attn[tid * 65 + j] *= inv;
  }
  __syncthreads();
  for (int idx = tid; idx < 2048; idx += 256){
    int i = idx >> 5, d = idx & 31;
    float s = 0.f;
    for (int j = 0; j < 64; ++j) s += attn[i * 65 + j] * vr[j * 33 + d];
    outp[((long)b * 256 + head * 32 + d) * 64 + i] = s;
  }
}

// Final fused kernel: B = relu(v + xx_row(h) + xx_col(w)); proj GEMM (M=128,K=256);
// epilogue: h_sigmoid(scale*acc+bias) * qkv_pw -> out. Same async/double-buffer pipeline.
__global__ void __launch_bounds__(256) final_proj_gate(
    const unsigned short* __restrict__ Wproj,
    const float* __restrict__ sproj, const float* __restrict__ bproj,
    const float* __restrict__ qkvf, const float* __restrict__ xxr,
    const float* __restrict__ xxc, const float* __restrict__ pw,
    float* __restrict__ outp)
{
  __shared__ unsigned short Aa[2][64 * AST];
  __shared__ unsigned short Bt[2][64 * AST];
  const int b     = blockIdx.z;
  const int mBase = blockIdx.y * 64;
  const int nBase = blockIdx.x * 64;
  const int h0    = nBase >> 6;                 // 64-wide N tile => single image row
  const int tid   = threadIdx.x;
  const int lane  = tid & 31, wave = tid >> 5;
  const int mr    = wave & 3;
  const int ncA   = (wave >> 2) * 2;
  v8f acc0 = {0,0,0,0,0,0,0,0}, acc1 = {0,0,0,0,0,0,0,0};
  const float* vsrc = qkvf + ((long)b * 512 + 256) * 4096;
  const float* xr   = xxr + (long)b * 256 * 64;
  const float* xc   = xxc + (long)b * 256 * 64;

  const int ar = tid >> 2;
  const int ac = (tid & 3) * 8;
  const unsigned aLds[2] = {
    lds_off_of(&Aa[0][0]) + (unsigned)(ar * AST + ac) * 2u,
    lds_off_of(&Aa[1][0]) + (unsigned)(ar * AST + ac) * 2u };
  const unsigned short* aG = Wproj + (long)(mBase + ar) * 256 + ac;

  auto stageA = [&](int kt, int buf){
    async_load_b128(aLds[buf], (const void*)(aG + kt));
  };
  auto stageB = [&](int kt, int buf){
    unsigned short* Bd = Bt[buf];
    for (int idx = tid; idx < 2048; idx += 256){
      int kk = idx >> 6, j = idx & 63;
      int kg = kt + kk;
      float f = vsrc[(long)kg * 4096 + nBase + j] + xr[kg * 64 + h0] + xc[kg * 64 + j];
      Bd[j * AST + kk] = f32_to_bf16(fmaxf(f, 0.f));
    }
  };

  stageA(0, 0);
  stageB(0, 0);
  wait_async0();
  __syncthreads();
  for (int s = 0; s < 8; ++s){
    const int cur = s & 1;
    if (s + 1 < 8){
      stageA((s + 1) << 5, cur ^ 1);
      stageB((s + 1) << 5, cur ^ 1);
    }
    wmma_step(Aa[cur], Bt[cur], mr, ncA, lane, acc0, acc1);
    wait_async0();
    __syncthreads();
  }

  const int nl   = lane & 15;
  const int mOff = (lane < 16) ? 0 : 8;
#pragma unroll
  for (int r = 0; r < 8; ++r){
    int m = mBase + mr * 16 + r + mOff;
    float s = sproj[m], bi = bproj[m];
    long base = ((long)b * 128 + m) * 4096;
    int n0 = nBase + ncA * 16 + nl;
    float p0 = acc0[r] * s + bi;
    float g0 = fminf(fmaxf(p0 + 3.f, 0.f), 6.f) * (1.f / 6.f);
    outp[base + n0] = g0 * pw[base + n0];
    float p1 = acc1[r] * s + bi;
    float g1 = fminf(fmaxf(p1 + 3.f, 0.f), 6.f) * (1.f / 6.f);
    outp[base + n0 + 16] = g1 * pw[base + n0 + 16];
  }
}

extern "C" void kernel_launch(void* const* d_in, const int* in_sizes, int n_in,
                              void* d_out, int out_size, void* d_ws, size_t ws_size,
                              hipStream_t stream)
{
  (void)in_sizes; (void)n_in; (void)out_size; (void)ws_size;
  const float* x    = (const float*)d_in[0];
  const float* wq   = (const float*)d_in[1];
  const float* sq   = (const float*)d_in[2];
  const float* bq   = (const float*)d_in[3];
  const float* wk   = (const float*)d_in[4];
  const float* sk   = (const float*)d_in[5];
  const float* bk   = (const float*)d_in[6];
  const float* wv   = (const float*)d_in[7];
  const float* sv   = (const float*)d_in[8];
  const float* bv   = (const float*)d_in[9];
  const float* wdw  = (const float*)d_in[10];
  const float* sdw  = (const float*)d_in[11];
  const float* bdw  = (const float*)d_in[12];
  const float* wpw  = (const float*)d_in[13];
  const float* spw  = (const float*)d_in[14];
  const float* bpw  = (const float*)d_in[15];
  const float* wrow = (const float*)d_in[16];
  const float* srow = (const float*)d_in[17];
  const float* brow = (const float*)d_in[18];
  const float* wcol = (const float*)d_in[19];
  const float* scol = (const float*)d_in[20];
  const float* bcol = (const float*)d_in[21];
  const float* wprj = (const float*)d_in[22];
  const float* sprj = (const float*)d_in[23];
  const float* bprj = (const float*)d_in[24];
  const float* pe_rq= (const float*)d_in[25];
  const float* pe_rk= (const float*)d_in[26];
  const float* pe_cq= (const float*)d_in[27];
  const float* pe_ck= (const float*)d_in[28];
  float* outp = (float*)d_out;

  char* base = (char*)d_ws;
  size_t off = 0;
  auto take = [&](size_t bytes) -> void* {
    off = (off + 255) & ~(size_t)255;
    void* p = base + off;
    off += bytes;
    return p;
  };
  float*          qkvf   = (float*)take((size_t)16*512*4096*4);          // q|k|v f32
  unsigned short* dwb    = (unsigned short*)take((size_t)16*512*4096*2); // dw+relu bf16
  float*          qkv_pw = (float*)take((size_t)16*128*4096*4);
  float*          mrow   = (float*)take((size_t)16*512*64*4);
  float*          mcol   = (float*)take((size_t)16*512*64*4);
  float*          xxr_pre= (float*)take((size_t)16*256*64*4);
  float*          xxc_pre= (float*)take((size_t)16*256*64*4);
  float*          xx_row = (float*)take((size_t)16*256*64*4);
  float*          xx_col = (float*)take((size_t)16*256*64*4);
  unsigned short* Wqkv   = (unsigned short*)take(512*128*2);
  unsigned short* Wpw    = (unsigned short*)take(128*512*2);
  unsigned short* Wrow   = (unsigned short*)take(256*256*2);
  unsigned short* Wcol   = (unsigned short*)take(256*256*2);
  unsigned short* Wproj  = (unsigned short*)take(128*256*2);
  float*          sqkv   = (float*)take(512*4);
  float*          bqkv   = (float*)take(512*4);
  float*          peqr   = (float*)take(128*64*4);
  float*          pekr   = (float*)take(128*64*4);
  float*          peqc   = (float*)take(128*64*4);
  float*          pekc   = (float*)take(128*64*4);

  pack_kernel<<<1282, 256, 0, stream>>>(wq, wk, wv, sq, bq, sk, bk, sv, bv,
                                        wpw, wrow, wcol, wprj,
                                        pe_rq, pe_rk, pe_cq, pe_ck,
                                        Wqkv, Wpw, Wrow, Wcol, Wproj,
                                        sqkv, bqkv, peqr, pekr, peqc, pekc);

  // q,k,v 1x1 conv: M=512, K=128, N=4096 per batch
  wmma_gemm<0><<<dim3(64, 8, 16), 256, 0, stream>>>(Wqkv, sqkv, bqkv, (const void*)x, qkvf,
                                                    512, 128, 4096,
                                                    (long)128*4096, (long)512*4096);
  means_kernel<<<16*512, 256, 0, stream>>>(qkvf, mrow, mcol);
  dwconv_relu_bf16<<<16*512, 256, 0, stream>>>(qkvf, wdw, sdw, bdw, dwb);
  // pwconv: M=128, K=512, N=4096 (B already bf16)
  wmma_gemm<2><<<dim3(64, 2, 16), 256, 0, stream>>>(Wpw, spw, bpw, (const void*)dwb, qkv_pw,
                                                    128, 512, 4096,
                                                    (long)512*4096, (long)128*4096);
  axial_attn<<<dim3(128, 2, 1), 256, 0, stream>>>(mrow, mcol, peqr, pekr, peqc, pekc,
                                                  xxr_pre, xxc_pre);
  // row/col 1x1 convs with fused ReLU on B: M=256, K=256, N=64
  wmma_gemm<1><<<dim3(1, 4, 16), 256, 0, stream>>>(Wrow, srow, brow, (const void*)xxr_pre, xx_row,
                                                   256, 256, 64, (long)256*64, (long)256*64);
  wmma_gemm<1><<<dim3(1, 4, 16), 256, 0, stream>>>(Wcol, scol, bcol, (const void*)xxc_pre, xx_col,
                                                   256, 256, 64, (long)256*64, (long)256*64);
  // fused: relu(v + row + col) -> proj GEMM -> h_sigmoid gate * qkv_pw
  final_proj_gate<<<dim3(64, 2, 16), 256, 0, stream>>>(Wproj, sprj, bprj, qkvf,
                                                       xx_row, xx_col, qkv_pw, outp);
}
